// MultiHeadSelfAttention_14989435863585
// MI455X (gfx1250) — matrix operand order repaired and checked by reading, not compiled, not run on hardware
//
#include <hip/hip_runtime.h>
#include <hip/hip_bf16.h>
#include <math.h>

// ---------------------------------------------------------------------------
// MHSA block for MI455X (gfx1250, wave32, WMMA 16x16x32 f16 -> f32 accum)
// ---------------------------------------------------------------------------

typedef __attribute__((ext_vector_type(16))) _Float16 v16h;
typedef __attribute__((ext_vector_type(8)))  _Float16 v8h;
typedef __attribute__((ext_vector_type(4)))  _Float16 v4h;
typedef __attribute__((ext_vector_type(8)))  float    v8f;
typedef __attribute__((ext_vector_type(4)))  float    v4f;

constexpr int   kB     = 4;
constexpr int   kN     = 2048;
constexpr int   kC     = 1024;
constexpr int   kH     = 8;
constexpr int   kHD    = 128;     // C / h
constexpr int   kM     = kB * kN; // 8192 total rows
constexpr int   k3C    = 3 * kC;  // 3072
constexpr float kScale = 0.03125f;   // C^-0.5 (full C, per reference)
constexpr float kEps   = 1e-3f;

__device__ __forceinline__ v8f v8f_zero() {
  v8f z;
#pragma unroll
  for (int i = 0; i < 8; ++i) z[i] = 0.0f;
  return z;
}

__device__ __forceinline__ v8f wmma_f16(v16h a, v16h b, v8f c) {
  // v_wmma_f32_16x16x32_f16  (neg_a, A, neg_b, B, c_mod, C, reuse_a, reuse_b)
  return __builtin_amdgcn_wmma_f32_16x16x32_f16(false, a, false, b, (short)0, c,
                                                false, false);
}

// A fragment 16x32 f16 from row-major LDS tile (lda in halves, multiple of 8).
// ISA layout: lanes 0-15 row M=lane, k-base 0; lanes 16-31 same rows, k-base 8.
// elements 0..7 = K kb+0..7, elements 8..15 = K 16+kb+0..7.
__device__ __forceinline__ v16h lds_a_frag(const _Float16* A, int row0, int k0,
                                           int lda, int lane) {
  const int r  = lane & 15;
  const int kb = (lane >> 4) << 3;  // 0 or 8
  const _Float16* p = A + (row0 + r) * lda + k0 + kb;
  v8h lo = *(const v8h*)(p);
  v8h hi = *(const v8h*)(p + 16);
  v16h f;
#pragma unroll
  for (int i = 0; i < 8; ++i) { f[i] = lo[i]; f[i + 8] = hi[i]; }
  return f;
}

// B fragment 32x16 f16 from column-major ("Bt") LDS tile: Bt[col][k], ldb mult 8.
// lanes 0-15: col=lane, K 0..7 and 16..23; lanes 16-31: K 8..15 and 24..31.
__device__ __forceinline__ v16h lds_b_frag(const _Float16* Bt, int col0, int k0,
                                           int ldb, int lane) {
  const int c  = lane & 15;
  const int kb = (lane >> 4) << 3;  // 0 or 8
  const _Float16* p = Bt + (col0 + c) * ldb + k0 + kb;
  v8h lo = *(const v8h*)(p);
  v8h hi = *(const v8h*)(p + 16);
  v16h f;
#pragma unroll
  for (int i = 0; i < 8; ++i) { f[i] = lo[i]; f[i + 8] = hi[i]; }
  return f;
}

// ---------------------------------------------------------------------------
// Kernel 1: LayerNorm over C, write f16 activations.
// One block (256 threads) per row; each thread handles 4 contiguous floats.
// ---------------------------------------------------------------------------
__global__ __launch_bounds__(256) void ln_f16_kernel(
    const float* __restrict__ x, const float* __restrict__ gamma,
    const float* __restrict__ beta, _Float16* __restrict__ xn) {
  __shared__ float ssum[8], ssum2[8], sstat[2];
  const int row  = blockIdx.x;
  const int t    = threadIdx.x;
  const int lane = t & 31;
  const int w    = t >> 5;
  const float* xr = x + (size_t)row * kC;
  v4f xv = *(const v4f*)(xr + t * 4);
  float s  = xv[0] + xv[1] + xv[2] + xv[3];
  float s2 = xv[0] * xv[0] + xv[1] * xv[1] + xv[2] * xv[2] + xv[3] * xv[3];
#pragma unroll
  for (int m = 16; m >= 1; m >>= 1) {
    s  += __shfl_xor(s, m, 32);
    s2 += __shfl_xor(s2, m, 32);
  }
  if (lane == 0) { ssum[w] = s; ssum2[w] = s2; }
  __syncthreads();
  if (t == 0) {
    float a = 0.0f, b = 0.0f;
#pragma unroll
    for (int i = 0; i < 8; ++i) { a += ssum[i]; b += ssum2[i]; }
    float mu  = a / (float)kC;
    float var = b / (float)kC - mu * mu;
    sstat[0] = mu;
    sstat[1] = rsqrtf(var + kEps);
  }
  __syncthreads();
  const float mu = sstat[0], rstd = sstat[1];
  v4h o;
#pragma unroll
  for (int i = 0; i < 4; ++i) {
    int c = t * 4 + i;
    float v = (xv[i] - mu) * rstd * gamma[c] + beta[c];
    o[i] = (_Float16)v;
  }
  *(v4h*)(xn + (size_t)row * kC + t * 4) = o;
}

// ---------------------------------------------------------------------------
// GEMM: Y[kM x NCOLS] = A_f16[kM x 1024] * W_f32[1024 x NCOLS]
// Block 256 threads (8 waves), tile 128x128, BK=32.
// Wave = 32x64 tile: 2 A-frags + 4 B-frags -> 8 WMMA per K-step.
// EPI==0: scatter f16 into q/k/v [3][B][h][N][hd]   (qkv projection)
// EPI==1: f32 out = acc + bias[col] + xres[row,col]  (output projection)
// ---------------------------------------------------------------------------
template <int NCOLS, int EPI>
__global__ __launch_bounds__(256) void gemm_kernel(
    const _Float16* __restrict__ A, const float* __restrict__ W,
    _Float16* __restrict__ qkv_out, const float* __restrict__ bias,
    const float* __restrict__ xres, float* __restrict__ yout) {
  constexpr int BM = 128, BN = 128, BK = 32, LDA = 40, LDB = 40;
  __shared__ _Float16 As[BM * LDA];    // A tile row-major [m][k]
  __shared__ _Float16 Bts[BN * LDB];   // W tile transposed [n][k]

  const int t    = threadIdx.x;
  const int lane = t & 31;
  const int w    = t >> 5;
  const int wr   = w >> 1;  // 0..3 -> 32-row strip
  const int wc   = w & 1;   // 0..1 -> 64-col strip
  const int m0   = blockIdx.y * BM;
  const int n0   = blockIdx.x * BN;

  v8f acc[2][4];
#pragma unroll
  for (int i = 0; i < 2; ++i)
#pragma unroll
    for (int j = 0; j < 4; ++j) acc[i][j] = v8f_zero();

  for (int kb = 0; kb < kC; kb += BK) {
    __syncthreads();
    // A tile 128x32 f16: 512 chunks of 8 halves, 2 per thread
#pragma unroll
    for (int it = 0; it < 2; ++it) {
      int id = t + it * 256;
      int r  = id >> 2;
      int cc = (id & 3) << 3;
      v8h av = *(const v8h*)(A + (size_t)(m0 + r) * kC + kb + cc);
      *(v8h*)(As + r * LDA + cc) = av;
    }
    // W tile 32x128 f32 -> f16, stored transposed Bt[col][k]; 4 float4/thread
#pragma unroll
    for (int it = 0; it < 4; ++it) {
      int id = t + it * 256;
      int kk = id >> 5;          // 0..31
      int cc = (id & 31) << 2;   // 0..124
      v4f bv = *(const v4f*)(W + (size_t)(kb + kk) * NCOLS + n0 + cc);
#pragma unroll
      for (int j = 0; j < 4; ++j) Bts[(cc + j) * LDB + kk] = (_Float16)bv[j];
    }
    __syncthreads();
    v16h a0 = lds_a_frag(As, wr * 32, 0, LDA, lane);
    v16h a1 = lds_a_frag(As, wr * 32 + 16, 0, LDA, lane);
#pragma unroll
    for (int j = 0; j < 4; ++j) {
      v16h bf = lds_b_frag(Bts, wc * 64 + j * 16, 0, LDB, lane);
      acc[0][j] = wmma_f16(a0, bf, acc[0][j]);
      acc[1][j] = wmma_f16(a1, bf, acc[1][j]);
    }
  }

  // Epilogue. C layout: element r -> row 8*(lane>=16)+r, col lane&15.
  const int cr = lane & 15;
  const int r8 = (lane >> 4) << 3;
#pragma unroll
  for (int i = 0; i < 2; ++i)
#pragma unroll
    for (int j = 0; j < 4; ++j) {
      const int colg = n0 + wc * 64 + j * 16 + cr;
#pragma unroll
      for (int r = 0; r < 8; ++r) {
        const int rowg = m0 + wr * 32 + i * 16 + r8 + r;
        const float v = acc[i][j][r];
        if (EPI == 0) {
          const int qi   = colg >> 10;   // qkv index
          const int rem  = colg & 1023;
          const int head = rem >> 7;
          const int d    = rem & 127;
          const int b    = rowg >> 11;
          const int n    = rowg & 2047;
          size_t off = (size_t)qi * (size_t)kM * kC +
                       (((size_t)(b * kH + head) * kN + n) * kHD + d);
          qkv_out[off] = (_Float16)v;
        } else {
          size_t off = (size_t)rowg * NCOLS + colg;
          yout[off] = v + bias[colg] + xres[off];
        }
      }
    }
}

// ---------------------------------------------------------------------------
// Kernel 3: flash attention. One block (128 threads, 4 waves) per
// (b*h, 64-row q tile). Each wave owns 16 q rows; online softmax in registers
// via shfl_xor (masks 1/2/4/8 stay inside the 16-lane row-replication group).
// ---------------------------------------------------------------------------
__global__ __launch_bounds__(128) void attn_kernel(
    const _Float16* __restrict__ qkv, _Float16* __restrict__ attn_out) {
  constexpr int TR = 64, TC = 64;
  constexpr int LDQ = 136, LDK = 136, LDV = 72, LDP = 72;  // halves, mult of 8
  __shared__ _Float16 Qs[TR * LDQ];   // Q tile row-major [row][d]
  __shared__ _Float16 Ks[TC * LDK];   // K tile row-major [j][d] == Bt for QK^T
  __shared__ _Float16 Vt[kHD * LDV];  // V transposed [d][j] == Bt for P@V
  __shared__ _Float16 Ps[TR * LDP];   // probabilities, per-wave 16-row strips

  const int t    = threadIdx.x;  // 0..127
  const int lane = t & 31;
  const int w    = t >> 5;       // 0..3
  const int bh   = blockIdx.y;   // b*8 + head
  const int row0 = blockIdx.x * TR;

  const _Float16* qb = qkv + (size_t)bh * kN * kHD;
  const _Float16* kb = qkv + (size_t)kM * kC + (size_t)bh * kN * kHD;
  const _Float16* vb = qkv + 2 * (size_t)kM * kC + (size_t)bh * kN * kHD;

  // Load Q tile (64x128): 1024 chunks of 8 halves, 8 per thread
#pragma unroll
  for (int it = 0; it < 8; ++it) {
    int id = t + it * 128;
    int r  = id >> 4;
    int cc = (id & 15) << 3;
    v8h qv = *(const v8h*)(qb + (size_t)(row0 + r) * kHD + cc);
    *(v8h*)(Qs + r * LDQ + cc) = qv;
  }

  const int cr = lane & 15;
  const int r8 = (lane >> 4) << 3;

  v8f o[8];
#pragma unroll
  for (int dt = 0; dt < 8; ++dt) o[dt] = v8f_zero();
  float m_r[8], l_r[8];
#pragma unroll
  for (int r = 0; r < 8; ++r) { m_r[r] = -3.0e38f; l_r[r] = 0.0f; }

  for (int j0 = 0; j0 < kN; j0 += TC) {
    __syncthreads();
    // K tile row-major
#pragma unroll
    for (int it = 0; it < 8; ++it) {
      int id = t + it * 128;
      int r  = id >> 4;
      int cc = (id & 15) << 3;
      v8h kv = *(const v8h*)(kb + (size_t)(j0 + r) * kHD + cc);
      *(v8h*)(Ks + r * LDK + cc) = kv;
    }
    // V tile transposed into Vt[d][j]
#pragma unroll
    for (int it = 0; it < 8; ++it) {
      int id = t + it * 128;
      int r  = id >> 4;
      int cc = (id & 15) << 3;
      v8h vv = *(const v8h*)(vb + (size_t)(j0 + r) * kHD + cc);
#pragma unroll
      for (int e = 0; e < 8; ++e) Vt[(cc + e) * LDV + r] = vv[e];
    }
    __syncthreads();

    // S = Q K^T : wave rows [w*16, w*16+16), 4 col tiles, K-dim 128
    v8f s[4];
#pragma unroll
    for (int j = 0; j < 4; ++j) s[j] = v8f_zero();
#pragma unroll
    for (int kk = 0; kk < kHD; kk += 32) {
      v16h af = lds_a_frag(Qs, w * 16, kk, LDQ, lane);
#pragma unroll
      for (int j = 0; j < 4; ++j) {
        v16h bf = lds_b_frag(Ks, j * 16, kk, LDK, lane);
        s[j] = wmma_f16(af, bf, s[j]);
      }
    }

    // Online softmax per accumulator row; write P (f16) in A-layout strip.
#pragma unroll
    for (int r = 0; r < 8; ++r) {
      float mx = fmaxf(fmaxf(s[0][r], s[1][r]), fmaxf(s[2][r], s[3][r]));
      mx *= kScale;
#pragma unroll
      for (int msk = 1; msk <= 8; msk <<= 1)
        mx = fmaxf(mx, __shfl_xor(mx, msk, 32));
      const float mold = m_r[r];
      const float mn   = fmaxf(mold, mx);
      const float al   = __expf(mold - mn);
      float ls = 0.0f;
      _Float16* pr = Ps + (w * 16 + r8 + r) * LDP;
#pragma unroll
      for (int j = 0; j < 4; ++j) {
        float p = __expf(s[j][r] * kScale - mn);
        ls += p;
        pr[j * 16 + cr] = (_Float16)p;
      }
#pragma unroll
      for (int msk = 1; msk <= 8; msk <<= 1) ls += __shfl_xor(ls, msk, 32);
      m_r[r] = mn;
      l_r[r] = l_r[r] * al + ls;
#pragma unroll
      for (int dt = 0; dt < 8; ++dt) o[dt][r] *= al;  // rescale O rows
    }

    // O += P[16x64] @ V[64x128] (same-wave LDS write->read, in-order DS)
#pragma unroll
    for (int kk = 0; kk < TC; kk += 32) {
      v16h af = lds_a_frag(Ps, w * 16, kk, LDP, lane);
#pragma unroll
      for (int dt = 0; dt < 8; ++dt) {
        v16h bf = lds_b_frag(Vt, dt * 16, kk, LDV, lane);
        o[dt] = wmma_f16(af, bf, o[dt]);
      }
    }
  }

  // Finalize: divide by l, write back to [b, n, head*128 + d] f16.
  const int b    = bh >> 3;
  const int head = bh & 7;
  float inv_r[8];
#pragma unroll
  for (int r = 0; r < 8; ++r) inv_r[r] = 1.0f / l_r[r];
#pragma unroll
  for (int dt = 0; dt < 8; ++dt)
#pragma unroll
    for (int r = 0; r < 8; ++r) {
      const int grow = row0 + w * 16 + r8 + r;
      const int d    = dt * 16 + cr;
      size_t off = ((size_t)b * kN + grow) * kC + head * kHD + d;
      attn_out[off] = (_Float16)(o[dt][r] * inv_r[r]);
    }
}

// ---------------------------------------------------------------------------
// Host launcher. Workspace (f16): xn[8M], q/k/v[3*8M], attn_out[8M] = 80 MB.
// ---------------------------------------------------------------------------
extern "C" void kernel_launch(void* const* d_in, const int* in_sizes, int n_in,
                              void* d_out, int out_size, void* d_ws,
                              size_t ws_size, hipStream_t stream) {
  (void)in_sizes; (void)n_in; (void)out_size; (void)ws_size;
  const float* x     = (const float*)d_in[0];
  const float* gamma = (const float*)d_in[1];
  const float* beta  = (const float*)d_in[2];
  const float* w_qkv = (const float*)d_in[3];
  const float* w_out = (const float*)d_in[4];
  const float* b_out = (const float*)d_in[5];
  float* out = (float*)d_out;

  _Float16* wsh  = (_Float16*)d_ws;
  _Float16* xn   = wsh;                            // kM*kC
  _Float16* qkv  = wsh + (size_t)kM * kC;          // 3 * kM*kC
  _Float16* attn = wsh + 4 * (size_t)kM * kC;      // kM*kC

  // 1) LayerNorm -> f16
  ln_f16_kernel<<<kM, 256, 0, stream>>>(x, gamma, beta, xn);
  // 2) QKV projection (scatter into per-head layout), tile 128x128
  gemm_kernel<k3C, 0><<<dim3(k3C / 128, kM / 128), 256, 0, stream>>>(
      xn, w_qkv, qkv, nullptr, nullptr, nullptr);
  // 3) Flash attention per (b*h, 64-row tile)
  attn_kernel<<<dim3(kN / 64, kB * kH), 128, 0, stream>>>(qkv, attn);
  // 4) Output projection + bias + residual, tile 128x128
  gemm_kernel<kC, 1><<<dim3(kC / 128, kM / 128), 256, 0, stream>>>(
      attn, w_out, nullptr, b_out, x, out);
}
